// DiscreteSelectorTransform_214748365028
// MI455X (gfx1250) — matrix operand order repaired and checked by reading, not compiled, not run on hardware
//
#include <hip/hip_runtime.h>

// ---------------------------------------------------------------------------
// SelectedBijector forward on MI455X (gfx1250, wave32).
//   y:[B,512] f32, W:[8,512,1024] f32, b:[8,1024] f32, x:[B] int
//   out[i] = (x[i]==0) ? y[i] : y[i]*exp(ls_i) + shift_i
//   [shift|ls]_i = y[i] @ (W[x[i]] * trilmask) + b[x[i]]
// Strategy: counting-sort rows by expert, 16-row tiles per expert, bf16 hi/lo
// split (3x v_wmma_f32_16x16x32_bf16) emulating fp32, tril-mask chunk skipping,
// vectorized (b128) LDS staging, fused exp/FMA epilogue.
// ---------------------------------------------------------------------------

#define DD 512
#define TWO_D 1024

typedef __bf16 bf16_t;
typedef bf16_t v16bf __attribute__((ext_vector_type(16)));
typedef bf16_t v8bf  __attribute__((ext_vector_type(8)));
typedef float  v8f   __attribute__((ext_vector_type(8)));

// workspace layout (int32 units)
#define WS_COUNTS   0      // [8]
#define WS_POFF     8      // [9]  padded segment starts, expert 1..7, +end
#define WS_CURSOR   17     // +k for k=1..7  -> 18..24
#define WS_NTILES   25
#define WS_TILEEXP  32     // tile -> expert (<= B/16+7 entries)
#define WS_PERM     1088   // permuted row indices (B + pad)

__global__ void k_init(int* ws) {
  if (threadIdx.x < 32) ws[threadIdx.x] = 0;
}

__global__ void k_hist(const int* __restrict__ x, int* ws, int B) {
  int i = blockIdx.x * blockDim.x + threadIdx.x;
  if (i < B) {
    int k = x[i];
    if (k > 0) atomicAdd(&ws[WS_COUNTS + k], 1);
  }
}

__global__ void k_scan(int* ws) {
  if (threadIdx.x || blockIdx.x) return;
  int off = 0;
  for (int k = 1; k < 8; ++k) {
    ws[WS_POFF + k] = off;
    ws[WS_CURSOR + k] = off;
    int c = ws[WS_COUNTS + k];
    int tiles = (c + 15) >> 4;                 // 16-row aligned segments
    for (int tt = 0; tt < tiles; ++tt) ws[WS_TILEEXP + (off >> 4) + tt] = k;
    off += tiles << 4;
  }
  ws[WS_POFF + 8] = off;
  ws[WS_NTILES] = off >> 4;
}

__global__ void k_scatter(const int* __restrict__ x, int* ws, int B) {
  int i = blockIdx.x * blockDim.x + threadIdx.x;
  if (i < B) {
    int k = x[i];
    if (k > 0) {
      int pos = atomicAdd(&ws[WS_CURSOR + k], 1);
      ws[WS_PERM + pos] = i;
    }
  }
}

__global__ void k_pad(int* ws) {
  if (threadIdx.x || blockIdx.x) return;
  for (int k = 1; k < 8; ++k) {
    int s = ws[WS_POFF + k];
    int e = ws[WS_POFF + k + 1];
    int c = ws[WS_COUNTS + k];
    if (e > s + c) {
      int first = ws[WS_PERM + s];             // duplicate rows: benign stores
      for (int idx = s + c; idx < e; ++idx) ws[WS_PERM + idx] = first;
    }
  }
}

// identity expert: rows with x==0 are passed through
__global__ void k_identity(const float4* __restrict__ y4, const int* __restrict__ x,
                           float4* __restrict__ out4, int B) {
  int t = blockIdx.x * blockDim.x + threadIdx.x;
  int row = t >> 7;                            // D/4 = 128 float4 per row
  if (row < B && x[row] == 0) out4[t] = y4[t];
}

// main fused masked-GEMM + MAF epilogue
__launch_bounds__(256)
__global__ void k_gemm(const float* __restrict__ y, const float* __restrict__ W,
                       const float* __restrict__ b, const int* __restrict__ ws,
                       float* __restrict__ out) {
  const int t = blockIdx.x;
  if (t >= ws[WS_NTILES]) return;              // uniform early-exit, pre-barrier
  const int kexp = ws[WS_TILEEXP + t];
  const int j0   = blockIdx.y << 7;            // column-tile base in [0,512)
  const int tid  = threadIdx.x;
  const int lane = tid & 31;
  const int wv   = tid >> 5;                   // 8 waves x 16 cols = 128 cols
  const int row0 = t << 4;
  const int* perm = ws + WS_PERM;

  // LDS staging: y-tile (bf16 hi/lo) + W chunk transposed [col][k] (hi/lo)
  __shared__ __align__(16) bf16_t Ah[16][520]; // 520: bank-skew, 16B-multiple row
  __shared__ __align__(16) bf16_t Al[16][520];
  __shared__ __align__(16) bf16_t Bh[128][40]; // 40: 16B-aligned row stride
  __shared__ __align__(16) bf16_t Bl[128][40];

  // ---- stage A: 16 rows x 512 K of y, fp32 -> bf16 hi+lo, b128 DS stores ----
  {
    const int m = tid >> 4;                    // each thread: 32 contiguous K
    const int cbase = (tid & 15) << 5;
    const int sample = perm[row0 + m];
    const float4* src = (const float4*)&y[(long)sample * DD + cbase];
#pragma unroll
    for (int q = 0; q < 4; ++q) {
      float4 va = src[2 * q];
      float4 vb = src[2 * q + 1];
      float vals[8] = {va.x, va.y, va.z, va.w, vb.x, vb.y, vb.z, vb.w};
      v8bf hv, lv;
#pragma unroll
      for (int e = 0; e < 8; ++e) {
        bf16_t hi = (bf16_t)vals[e];
        hv[e] = hi;
        lv[e] = (bf16_t)(vals[e] - (float)hi);
      }
      *(v8bf*)&Ah[m][cbase + (q << 3)] = hv;
      *(v8bf*)&Al[m][cbase + (q << 3)] = lv;
    }
  }

  v8f acc0 = {};                               // shift half  (e = j)
  v8f acc1 = {};                               // log-scale half (e = j+512)
  const long wbase = (long)kexp * DD * TWO_D;
  const int mrow = lane & 15;
  const int coln = (wv << 4) + (lane & 15);
  const int koff = (lane < 16) ? 0 : 16;       // B frag: contiguous K run per lane

  // W staging role of this thread: one column, half of the K range
  const int scol  = tid & 127;
  const int sksel = (tid >> 7) << 4;           // 0 or 16
  const int sj    = j0 + scol;

  for (int kb = 0; kb < DD; kb += 32) {
    // tril mask: out col j depends on d > j; chunk dead if kb+31 <= j0
    if (kb + 31 <= j0) continue;

    __syncthreads();                           // covers A-stage (1st iter) + reuse

    // A fragments for this K chunk (contiguous K order, matching B):
    // lane<16: K = kb+[0..15]; lane>=16: K = kb+[16..31]
    const int c0 = kb + ((lane < 16) ? 0 : 16);
    union { v16bf v; v8bf h[2]; } ahi, alo;
    ahi.h[0] = *(const v8bf*)&Ah[mrow][c0];
    ahi.h[1] = *(const v8bf*)&Ah[mrow][c0 + 8];
    alo.h[0] = *(const v8bf*)&Al[mrow][c0];
    alo.h[1] = *(const v8bf*)&Al[mrow][c0 + 8];

    // prefetch next K chunk of W into L2 (global_prefetch_b8)
    {
      int dp = kb + 32 + (tid >> 3);
      if (dp < DD) {
        const float* p = &W[wbase + (long)dp * TWO_D + j0 + ((tid & 7) << 4)];
        __builtin_prefetch(p, 0, 1);
        __builtin_prefetch(p + DD, 0, 1);      // log-scale half
      }
    }

    for (int half = 0; half < 2; ++half) {
      if (half) __syncthreads();               // drain readers of previous half
      // stage W[kexp][kb+sksel .. +15][half*512 + j0+scol], masked, transposed.
      // 16 convergent strided b32 loads, then 4 b128 DS stores per thread.
      {
        const float* wcol =
            &W[wbase + (long)(kb + sksel) * TWO_D + (half << 9) + sj];
        float vals[16];
#pragma unroll
        for (int e = 0; e < 16; ++e) vals[e] = wcol[(long)e * TWO_D];
#pragma unroll
        for (int g = 0; g < 2; ++g) {
          v8bf hv, lv;
#pragma unroll
          for (int e = 0; e < 8; ++e) {
            float v = vals[(g << 3) + e];
            v = (kb + sksel + (g << 3) + e > sj) ? v : 0.0f;  // tril mask
            bf16_t hi = (bf16_t)v;
            hv[e] = hi;
            lv[e] = (bf16_t)(v - (float)hi);
          }
          *(v8bf*)&Bh[scol][sksel + (g << 3)] = hv;
          *(v8bf*)&Bl[scol][sksel + (g << 3)] = lv;
        }
      }
      __syncthreads();

      // B fragment: lane<16 holds K=0..15, lane>=16 holds K=16..31 (contiguous)
      union { v16bf v; v8bf h[2]; } bhi, blo;
      bhi.h[0] = *(const v8bf*)&Bh[coln][koff];
      bhi.h[1] = *(const v8bf*)&Bh[coln][koff + 8];
      blo.h[0] = *(const v8bf*)&Bl[coln][koff];
      blo.h[1] = *(const v8bf*)&Bl[coln][koff + 8];

      v8f acc = half ? acc1 : acc0;
      // fp32 emulation: hi*hi + lo*hi + hi*lo (drop lo*lo)
      acc = __builtin_amdgcn_wmma_f32_16x16x32_bf16(false, ahi.v, false, bhi.v,
                                                    (short)0, acc, false, false);
      acc = __builtin_amdgcn_wmma_f32_16x16x32_bf16(false, alo.v, false, bhi.v,
                                                    (short)0, acc, false, false);
      acc = __builtin_amdgcn_wmma_f32_16x16x32_bf16(false, ahi.v, false, blo.v,
                                                    (short)0, acc, false, false);
      if (half) acc1 = acc; else acc0 = acc;
    }
  }

  // ---- fused epilogue: u = y*exp(log_scale) + shift, scatter by perm ----
  const int jj = j0 + (wv << 4) + (lane & 15);
  const float bs  = b[kexp * TWO_D + jj];
  const float bls = b[kexp * TWO_D + DD + jj];
#pragma unroll
  for (int r = 0; r < 8; ++r) {
    int m = (lane < 16) ? r : (r + 8);         // f32 C/D tile layout
    int sample = perm[row0 + m];
    float shift = acc0[r] + bs;
    float ls    = acc1[r] + bls;
    float yv = y[(long)sample * DD + jj];
    out[(long)sample * DD + jj] = fmaf(yv, __expf(ls), shift);
  }
}

extern "C" void kernel_launch(void* const* d_in, const int* in_sizes, int n_in,
                              void* d_out, int out_size, void* d_ws, size_t ws_size,
                              hipStream_t stream) {
  const float* y = (const float*)d_in[0];
  const float* W = (const float*)d_in[1];
  const float* b = (const float*)d_in[2];
  const int*   x = (const int*)d_in[3];
  float* out = (float*)d_out;
  int* ws = (int*)d_ws;
  const int B = in_sizes[3];                   // x has B elements

  k_init<<<1, 32, 0, stream>>>(ws);
  k_hist<<<(B + 255) / 256, 256, 0, stream>>>(x, ws, B);
  k_scan<<<1, 1, 0, stream>>>(ws);
  k_scatter<<<(B + 255) / 256, 256, 0, stream>>>(x, ws, B);
  k_pad<<<1, 1, 0, stream>>>(ws);
  k_identity<<<(B * (DD / 4) + 255) / 256, 256, 0, stream>>>(
      (const float4*)y, x, (float4*)out, B);

  const int maxTiles = (B + 15) / 16 + 7;      // worst-case padded tile count
  dim3 grid(maxTiles, 4);                      // 4 x 128 cols; each does both halves
  k_gemm<<<grid, 256, 0, stream>>>(y, W, b, ws, out);
}
